// MultiHeadedAttentionAdaptiveSpan_75436805587619
// MI455X (gfx1250) — matrix operand order repaired and checked by reading, not compiled, not run on hardware
//
#include <hip/hip_runtime.h>
#include <stdint.h>

#define B_    4
#define T_    1024
#define FEAT_ 512
#define H_    8
#define DK_   64
#define SPAN_ 100
#define PADL_ 50

typedef __attribute__((ext_vector_type(16))) __bf16 v16bf;
typedef __attribute__((ext_vector_type(8)))  __bf16 v8bf;
typedef __attribute__((ext_vector_type(8)))  float  v8f;
typedef __attribute__((ext_vector_type(4)))  float  v4f;
typedef int v4i_gcc __attribute__((vector_size(16)));   // matches builtin prototype

#if defined(__HIP_DEVICE_COMPILE__) && \
    __has_builtin(__builtin_amdgcn_global_load_async_to_lds_b128) && \
    __has_builtin(__builtin_amdgcn_s_wait_asynccnt)
#define ASYNC_LDS 1
#else
#define ASYNC_LDS 0
#endif

#if ASYNC_LDS
static __device__ __forceinline__ void async_b128(void* lds, const void* g) {
    __builtin_amdgcn_global_load_async_to_lds_b128(
        (__attribute__((address_space(1))) v4i_gcc*)(g),
        (__attribute__((address_space(3))) v4i_gcc*)(lds),
        0, 0);
}
static __device__ __forceinline__ void wait_async() {
    __builtin_amdgcn_s_wait_asynccnt(0);
}
#endif

static __device__ __forceinline__ v16bf make16(v8bf lo, v8bf hi) {
    v16bf r;
#pragma unroll
    for (int i = 0; i < 8; ++i) { r[i] = lo[i]; r[8 + i] = hi[i]; }
    return r;
}

static __device__ __forceinline__ float swap16(float x) {
    // group-of-32 swizzle, xor=0x10 (SWAPX16): exchange lanes l <-> l^16
    return __int_as_float(__builtin_amdgcn_ds_swizzle(__float_as_int(x), 0x401F));
}

// ---------------------------------------------------------------------------
// one-time f32 -> bf16 cast (n multiple of 2048)
// ---------------------------------------------------------------------------
__global__ __launch_bounds__(256)
void cast_bf16_kernel(const float* __restrict__ in, __bf16* __restrict__ out)
{
    int i = (blockIdx.x * 256 + threadIdx.x) * 8;
    v4f f0 = *(const v4f*)(in + i);
    v4f f1 = *(const v4f*)(in + i + 4);
    v8bf o;
#pragma unroll
    for (int e = 0; e < 4; ++e) { o[e] = (__bf16)f0[e]; o[4 + e] = (__bf16)f1[e]; }
    *(v8bf*)&out[i] = o;
}

// ---------------------------------------------------------------------------
// one-time weight transpose+cast: in[K=512][N=512] f32 -> out[N][K] bf16
// ---------------------------------------------------------------------------
__global__ __launch_bounds__(256)
void transpose_cast_kernel(const float* __restrict__ in, __bf16* __restrict__ out)
{
    int idx = blockIdx.x * 256 + threadIdx.x;    // 512 * 64 threads total
    int n  = idx >> 6;
    int k0 = (idx & 63) * 8;
    v8bf o;
#pragma unroll
    for (int e = 0; e < 8; ++e) o[e] = (__bf16)in[(size_t)(k0 + e) * FEAT_ + n];
    *(v8bf*)&out[(size_t)n * FEAT_ + k0] = o;
}

// ---------------------------------------------------------------------------
// C[M x N] = (A[M x K] @ Wt[N x K]^T + bias[N]) * scale
// A, Wt bf16 (Wt pre-transposed so staging is a contiguous copy).
// Output: bf16 (Cb) or f32 (Cf).  128 threads (4 waves), 64x64 tile, k-step 32.
// ---------------------------------------------------------------------------
__global__ __launch_bounds__(128)
void gemm_bf16_kernel(const __bf16* __restrict__ A, const __bf16* __restrict__ Wt,
                      const float* __restrict__ bias,
                      float* __restrict__ Cf, __bf16* __restrict__ Cb,
                      int M, int N, int K, float scale)
{
    __shared__ __bf16 As[64 * 32];   // [row][k]
    __shared__ __bf16 Bs[64 * 32];   // [n][k]

    const int tid  = threadIdx.x;
    const int lane = tid & 31;
    const int wave = tid >> 5;
    const int hf   = lane >> 4;
    const int l16  = lane & 15;
    const int m0   = blockIdx.y * 64;
    const int n0   = blockIdx.x * 64;

    v8f acc[4] = {};

    for (int k0 = 0; k0 < K; k0 += 32) {
        __syncthreads();
#if ASYNC_LDS
#pragma unroll
        for (int j = 0; j < 2; ++j) {
            int g = tid * 2 + j;             // 256 granules of 16B per tile
            int row = g >> 2, part = g & 3;
            async_b128(&As[row * 32 + part * 8],
                       A  + (size_t)(m0 + row) * K + k0 + part * 8);
            async_b128(&Bs[row * 32 + part * 8],
                       Wt + (size_t)(n0 + row) * K + k0 + part * 8);
        }
        wait_async();
#else
#pragma unroll
        for (int j = 0; j < 2; ++j) {
            int g = tid * 2 + j;
            int row = g >> 2, part = g & 3;
            *(v8bf*)&As[row * 32 + part * 8] =
                *(const v8bf*)(A  + (size_t)(m0 + row) * K + k0 + part * 8);
            *(v8bf*)&Bs[row * 32 + part * 8] =
                *(const v8bf*)(Wt + (size_t)(n0 + row) * K + k0 + part * 8);
        }
#endif
        __syncthreads();

        int arow = wave * 16 + l16;
        v16bf af = make16(*(const v8bf*)&As[arow * 32 + hf * 8],
                          *(const v8bf*)&As[arow * 32 + 16 + hf * 8]);
        v16bf bf4[4];
#pragma unroll
        for (int nt = 0; nt < 4; ++nt) {
            int ncol = nt * 16 + l16;
            bf4[nt] = make16(*(const v8bf*)&Bs[ncol * 32 + hf * 8],
                             *(const v8bf*)&Bs[ncol * 32 + 16 + hf * 8]);
        }
#pragma unroll
        for (int nt = 0; nt < 4; ++nt)
            acc[nt] = __builtin_amdgcn_wmma_f32_16x16x32_bf16(
                false, af, false, bf4[nt], (short)0, acc[nt], false, false);
    }

#pragma unroll
    for (int nt = 0; nt < 4; ++nt) {
        int col = n0 + nt * 16 + l16;
        float bv = bias[col];
#pragma unroll
        for (int v = 0; v < 8; ++v) {
            int row = m0 + wave * 16 + v + hf * 8;
            float val = (acc[nt][v] + bv) * scale;
            if (Cb) Cb[(size_t)row * N + col] = (__bf16)val;
            else    Cf[(size_t)row * N + col] = val;
        }
    }
}

// ---------------------------------------------------------------------------
// Banded attention with adaptive span (bf16 Q/K/V in, bf16 X out).
// One wave handles 16 query rows of one (b,h); window slot s of row r maps to
// relative key c = r + s within 128 keys starting at kbase = t0 - 50.
// OOB keys are address-clamped: their scores get masked to -inf and their
// probabilities are exactly 0, so clamped data never contributes.
// ---------------------------------------------------------------------------
__global__ __launch_bounds__(32)
void attn_kernel(const __bf16* __restrict__ Qb, const __bf16* __restrict__ Kb,
                 const __bf16* __restrict__ Vb, const int* __restrict__ mask,
                 const float* __restrict__ span_size, __bf16* __restrict__ Xb)
{
    __shared__ __bf16 Qs[16 * 64];    // [row][d]
    __shared__ __bf16 Ks[128 * 64];   // [n][d]
    __shared__ __bf16 Vs[64 * 128];   // [d][n]  (transposed for B-fragments)
    __shared__ float  Ss[16 * 128];
    __shared__ __bf16 Ps[16 * 128];
#if ASYNC_LDS
    __shared__ __bf16 Vn[128 * 64];   // natural-layout V staging for async DMA
#endif

    const int lane = threadIdx.x;
    const int hf   = lane >> 4;
    const int l16  = lane & 15;

    const int blk = blockIdx.x;
    const int t0  = (blk & 63) * 16;
    const int h   = (blk >> 6) & 7;
    const int b   = blk >> 9;
    const int kbase = t0 - PADL_;

#if ASYNC_LDS
    // ---- async DMA staging: Q (4 x b128), K + V (32 x 2 x b128) ----
#pragma unroll
    for (int kk = 0; kk < 4; ++kk) {
        int g = lane + kk * 32;
        int row = g >> 3, part = g & 7;
        async_b128(&Qs[row * 64 + part * 8],
                   Qb + (size_t)(b * T_ + t0 + row) * FEAT_ + h * DK_ + part * 8);
    }
#pragma unroll
    for (int kk = 0; kk < 32; ++kk) {
        int g = lane + kk * 32;
        int n = g >> 3, part = g & 7;
        int kt = kbase + n;
        kt = kt < 0 ? 0 : (kt >= T_ ? T_ - 1 : kt);
        size_t src = (size_t)(b * T_ + kt) * FEAT_ + h * DK_ + part * 8;
        async_b128(&Ks[n * 64 + part * 8], Kb + src);
        async_b128(&Vn[n * 64 + part * 8], Vb + src);
    }
    wait_async();
    __syncthreads();
    // ---- LDS transpose Vn[n][d] -> Vs[d][n] ----
#pragma unroll
    for (int kk = 0; kk < 4; ++kk) {
        int n = lane + kk * 32;
#pragma unroll
        for (int j = 0; j < 8; ++j) {
            v8bf v = *(const v8bf*)&Vn[n * 64 + j * 8];
#pragma unroll
            for (int e = 0; e < 8; ++e) Vs[(j * 8 + e) * 128 + n] = v[e];
        }
    }
    __syncthreads();
#else
    // ---- fallback: manual staging ----
    {
        int row = lane >> 1;
        int c0  = (lane & 1) * 32;
        const __bf16* src = Qb + (size_t)(b * T_ + t0 + row) * FEAT_ + h * DK_ + c0;
#pragma unroll
        for (int j = 0; j < 4; ++j)
            *(v8bf*)&Qs[row * 64 + c0 + j * 8] = *(const v8bf*)(src + j * 8);
    }
#pragma unroll
    for (int kk = 0; kk < 4; ++kk) {
        int n  = lane + kk * 32;
        int kt = kbase + n;
        kt = kt < 0 ? 0 : (kt >= T_ ? T_ - 1 : kt);
        const __bf16* ksrc = Kb + (size_t)(b * T_ + kt) * FEAT_ + h * DK_;
        const __bf16* vsrc = Vb + (size_t)(b * T_ + kt) * FEAT_ + h * DK_;
#pragma unroll
        for (int j = 0; j < 8; ++j)
            *(v8bf*)&Ks[n * 64 + j * 8] = *(const v8bf*)(ksrc + j * 8);
#pragma unroll
        for (int j = 0; j < 8; ++j) {
            v8bf v = *(const v8bf*)(vsrc + j * 8);
#pragma unroll
            for (int e = 0; e < 8; ++e) Vs[(j * 8 + e) * 128 + n] = v[e];
        }
    }
    __syncthreads();
#endif

    // ---- S(16x128) = Q(16x64) @ Kwin^T(64x128) ----
    v16bf a0 = make16(*(const v8bf*)&Qs[l16 * 64 + 0  + hf * 8],
                      *(const v8bf*)&Qs[l16 * 64 + 16 + hf * 8]);
    v16bf a1 = make16(*(const v8bf*)&Qs[l16 * 64 + 32 + hf * 8],
                      *(const v8bf*)&Qs[l16 * 64 + 48 + hf * 8]);
#pragma unroll
    for (int nt = 0; nt < 8; ++nt) {
        int n = nt * 16 + l16;
        v16bf b0 = make16(*(const v8bf*)&Ks[n * 64 + hf * 8],
                          *(const v8bf*)&Ks[n * 64 + 16 + hf * 8]);
        v16bf b1 = make16(*(const v8bf*)&Ks[n * 64 + 32 + hf * 8],
                          *(const v8bf*)&Ks[n * 64 + 48 + hf * 8]);
        v8f acc = {};
        acc = __builtin_amdgcn_wmma_f32_16x16x32_bf16(
            false, a0, false, b0, (short)0, acc, false, false);
        acc = __builtin_amdgcn_wmma_f32_16x16x32_bf16(
            false, a1, false, b1, (short)0, acc, false, false);
#pragma unroll
        for (int v = 0; v < 8; ++v)
            Ss[(v + hf * 8) * 128 + n] = acc[v];
    }
    __syncthreads();

    // ---- softmax + span soft-mask + renorm; lane pair (l, l^16) owns row l16 ----
    {
        const int r = l16;
        const int cbase = hf * 64;
        const float span100 = span_size[h] * 100.0f;

        float sv[64];
#pragma unroll
        for (int j4 = 0; j4 < 16; ++j4) {
            v4f v = *(const v4f*)&Ss[r * 128 + cbase + j4 * 4];
#pragma unroll
            for (int e = 0; e < 4; ++e) sv[j4 * 4 + e] = v[e];
        }

        float mx = -3.0e38f;
#pragma unroll
        for (int j = 0; j < 64; ++j) {
            int c  = cbase + j;
            int s  = c - r;
            int kt = kbase + c;
            bool ok = (s >= 0) && (s < SPAN_) && (kt >= 0) && (kt < T_)
                      && (mask[b * T_ + kt] != 0);
            sv[j] = ok ? sv[j] : -3.0e38f;
            mx = fmaxf(mx, sv[j]);
        }
        mx = fmaxf(mx, swap16(mx));

        float sum_e = 0.f, sum_es = 0.f;
        float pv[64];
#pragma unroll
        for (int j = 0; j < 64; ++j) {
            int s = (cbase + j) - r;
            float e = (sv[j] > -1.0e38f) ? __expf(sv[j] - mx) : 0.f;
            // template[s] = s - 99 ; soft = clip((template + span*100)/2 + 1, 0, 1)
            float soft = fminf(fmaxf(((float)(s - 99) + span100) * 0.5f + 1.0f, 0.f), 1.f);
            float p = e * soft;
            sum_e  += e;
            sum_es += p;
            pv[j] = p;
        }
        sum_e  += swap16(sum_e);
        sum_es += swap16(sum_es);
        // final = e*soft / (sum_es + 1e-8*sum_e) == (attn*soft)/(sum(attn*soft)+1e-8)
        float inv = 1.0f / (sum_es + 1e-8f * sum_e);

#pragma unroll
        for (int j8 = 0; j8 < 8; ++j8) {
            v8bf o;
#pragma unroll
            for (int e = 0; e < 8; ++e) o[e] = (__bf16)(pv[j8 * 8 + e] * inv);
            *(v8bf*)&Ps[r * 128 + cbase + j8 * 8] = o;
        }
    }
    __syncthreads();

    // ---- X(16x64) = P(16x128) @ Vwin(128x64) ----
    v8f xacc[4] = {};
#pragma unroll
    for (int ks = 0; ks < 4; ++ks) {
        v16bf af = make16(*(const v8bf*)&Ps[l16 * 128 + ks * 32 + hf * 8],
                          *(const v8bf*)&Ps[l16 * 128 + ks * 32 + 16 + hf * 8]);
        v16bf bfv[4];
#pragma unroll
        for (int dt = 0; dt < 4; ++dt) {
            int d = dt * 16 + l16;
            bfv[dt] = make16(*(const v8bf*)&Vs[d * 128 + ks * 32 + hf * 8],
                             *(const v8bf*)&Vs[d * 128 + ks * 32 + 16 + hf * 8]);
        }
#pragma unroll
        for (int dt = 0; dt < 4; ++dt)
            xacc[dt] = __builtin_amdgcn_wmma_f32_16x16x32_bf16(
                false, af, false, bfv[dt], (short)0, xacc[dt], false, false);
    }
#pragma unroll
    for (int dt = 0; dt < 4; ++dt) {
        int col = h * DK_ + dt * 16 + l16;
#pragma unroll
        for (int v = 0; v < 8; ++v) {
            int row = b * T_ + t0 + v + hf * 8;
            Xb[(size_t)row * FEAT_ + col] = (__bf16)xacc[dt][v];
        }
    }
}

// ---------------------------------------------------------------------------
extern "C" void kernel_launch(void* const* d_in, const int* in_sizes, int n_in,
                              void* d_out, int out_size, void* d_ws, size_t ws_size,
                              hipStream_t stream)
{
    const float* query = (const float*)d_in[0];
    const float* key   = (const float*)d_in[1];
    const float* value = (const float*)d_in[2];
    const int*   mask  = (const int*)  d_in[3];
    const float* Wq    = (const float*)d_in[4];
    const float* bq    = (const float*)d_in[5];
    const float* Wk    = (const float*)d_in[6];
    const float* bk    = (const float*)d_in[7];
    const float* Wv    = (const float*)d_in[8];
    const float* bv    = (const float*)d_in[9];
    const float* Wo    = (const float*)d_in[10];
    const float* bo    = (const float*)d_in[11];
    const float* span  = (const float*)d_in[12];

    const int M = B_ * T_;                 // 4096
    const int N = FEAT_;                   // 512
    const int K = FEAT_;                   // 512
    const size_t MN = (size_t)M * N;       // 2,097,152
    const size_t KN = (size_t)K * N;       //   262,144

    __bf16* p   = (__bf16*)d_ws;
    __bf16* qin = p;           p += MN;
    __bf16* kin = p;           p += MN;
    __bf16* vin = p;           p += MN;
    __bf16* wqt = p;           p += KN;
    __bf16* wkt = p;           p += KN;
    __bf16* wvt = p;           p += KN;
    __bf16* wot = p;           p += KN;
    __bf16* Qb  = p;           p += MN;
    __bf16* Kb  = p;           p += MN;
    __bf16* Vb  = p;           p += MN;
    __bf16* Xb  = p;

    // one-time casts / weight transposes
    cast_bf16_kernel<<<(int)(MN / 2048), 256, 0, stream>>>(query, qin);
    cast_bf16_kernel<<<(int)(MN / 2048), 256, 0, stream>>>(key,   kin);
    cast_bf16_kernel<<<(int)(MN / 2048), 256, 0, stream>>>(value, vin);
    transpose_cast_kernel<<<128, 256, 0, stream>>>(Wq, wqt);
    transpose_cast_kernel<<<128, 256, 0, stream>>>(Wk, wkt);
    transpose_cast_kernel<<<128, 256, 0, stream>>>(Wv, wvt);
    transpose_cast_kernel<<<128, 256, 0, stream>>>(Wo, wot);

    dim3 gblk(128);
    dim3 ggrd(N / 64, M / 64);

    // projections (fold 1/sqrt(DK)=0.125 into Q), bf16 outputs
    gemm_bf16_kernel<<<ggrd, gblk, 0, stream>>>(qin, wqt, bq, nullptr, Qb, M, N, K, 0.125f);
    gemm_bf16_kernel<<<ggrd, gblk, 0, stream>>>(kin, wkt, bk, nullptr, Kb, M, N, K, 1.0f);
    gemm_bf16_kernel<<<ggrd, gblk, 0, stream>>>(vin, wvt, bv, nullptr, Vb, M, N, K, 1.0f);

    // banded attention: one wave per (b, h, 16-row tile)
    attn_kernel<<<B_ * H_ * (T_ / 16), 32, 0, stream>>>(Qb, Kb, Vb, mask, span, Xb);

    // output projection, f32 output
    gemm_bf16_kernel<<<ggrd, gblk, 0, stream>>>(Xb, wot, bo, (float*)d_out, nullptr, M, N, K, 1.0f);
}
